// Speller_82815559401846
// MI455X (gfx1250) — matrix operand order repaired and checked by reading, not compiled, not run on hardware
//
#include <hip/hip_runtime.h>

typedef __attribute__((ext_vector_type(16))) __bf16 v16bf;
typedef __attribute__((ext_vector_type(8)))  __bf16 v8bf;
typedef __attribute__((ext_vector_type(8)))  float  v8f;

#define B_    64
#define S_    1600
#define T_    256
#define ENC_  512
#define P_    256
#define E_    512
#define H_    512
#define V_    1000
#define G_    2048   /* 4*H */
#define VPAD_ 1008   /* V padded to multiple of 16 */

#define WMMA_BF16(a, b, c) \
  __builtin_amdgcn_wmma_f32_16x16x32_bf16(false, (a), false, (b), (short)0, (c), false, false)

// ---------------- fragment loaders (wave32 WMMA 16x16x32 bf16) ----------------
// A-matrix 16x32: lane<16 row=lane, K chunks {k0..k0+7} and {k0+16..k0+23};
// lane>=16 same rows, chunks shifted by +8.
__device__ __forceinline__ v16bf fragA_f32(const float* rowp, int k0, int lane) {
  const float* p = rowp + k0 + ((lane >> 4) << 3);
  v16bf a;
#pragma unroll
  for (int i = 0; i < 8; ++i) { a[i] = (__bf16)p[i]; a[i + 8] = (__bf16)p[i + 16]; }
  return a;
}
__device__ __forceinline__ v16bf fragA_bf16(const __bf16* rowp, int k0, int lane) {
  const __bf16* p = rowp + k0 + ((lane >> 4) << 3);
  v16bf a;
#pragma unroll
  for (int i = 0; i < 8; ++i) { a[i] = p[i]; a[i + 8] = p[i + 16]; }
  return a;
}
// B-matrix 32x16: lane holds column n=lane&15, K chunks {k0..k0+7} and
// {k0+16..k0+23}; lane>=16 chunks shifted by +8 (same order as A).
__device__ __forceinline__ v16bf fragB_bf16(const __bf16* rowp, int k0, int lane) {
  const __bf16* p = rowp + k0 + ((lane >> 4) << 3);
  v16bf b;
#pragma unroll
  for (int i = 0; i < 8; ++i) { b[i] = p[i]; b[i + 8] = p[i + 16]; }
  return b;
}

__device__ __forceinline__ float sigm(float x) { return 1.f / (1.f + __expf(-x)); }

// ---------------- prep kernels ----------------
__global__ void k_f32_to_bf16(const float* __restrict__ src, __bf16* __restrict__ dst,
                              int n, int ntot) {
  int i = blockIdx.x * 256 + threadIdx.x;
  if (i < ntot) dst[i] = (i < n) ? (__bf16)src[i] : (__bf16)0.f;
}
__global__ void k_zero_f32(float* p, int n) {
  int i = blockIdx.x * 256 + threadIdx.x;
  if (i < n) p[i] = 0.f;
}
__global__ void k_zero_bf16(__bf16* p, int n) {
  int i = blockIdx.x * 256 + threadIdx.x;
  if (i < n) p[i] = (__bf16)0.f;
}

// ---------------- key/val projection: [B*S,ENC] x [ENC,P] -> bf16 K/V (L2-resident) ----
__global__ __launch_bounds__(256) void k_proj(const float* __restrict__ enc,
                                              const __bf16* __restrict__ wk,
                                              const __bf16* __restrict__ wv,
                                              const float* __restrict__ bk,
                                              const float* __restrict__ bv,
                                              __bf16* __restrict__ key,
                                              __bf16* __restrict__ val) {
  int lane = threadIdx.x & 31, wave = threadIdx.x >> 5;
  int tile = blockIdx.x * 8 + wave;        // 102400 tiles total
  int m0 = (tile >> 4) * 16, n0 = (tile & 15) * 16;
  const float*  arow = enc + (size_t)(m0 + (lane & 15)) * ENC_;
  const __bf16* kb   = wk + (size_t)(n0 + (lane & 15)) * ENC_;
  const __bf16* vb   = wv + (size_t)(n0 + (lane & 15)) * ENC_;
  v8f ak = {0, 0, 0, 0, 0, 0, 0, 0}, av = {0, 0, 0, 0, 0, 0, 0, 0};
#pragma unroll 1
  for (int k0 = 0; k0 < ENC_; k0 += 32) {
    v16bf a = fragA_f32(arow, k0, lane);
    ak = WMMA_BF16(a, fragB_bf16(kb, k0, lane), ak);
    av = WMMA_BF16(a, fragB_bf16(vb, k0, lane), av);
  }
  int n = n0 + (lane & 15), mb = m0 + ((lane >> 4) << 3);
  float bku = bk[n], bvu = bv[n];
#pragma unroll
  for (int r = 0; r < 8; ++r) {
    key[(size_t)(mb + r) * P_ + n] = (__bf16)(ak[r] + bku);
    val[(size_t)(mb + r) * P_ + n] = (__bf16)(av[r] + bvu);
  }
}

// ---------------- shared LSTM epilogue: gates(LDS) -> c, h(bf16) ----------------
// gsh[gate][row16][col32] holds the block's 16x32 slice of all four gates.
__device__ __forceinline__ void lstm_epilogue(float (*gsh)[16][32], int m0, int cg,
                                              const float* __restrict__ bih,
                                              const float* __restrict__ bhh,
                                              float* __restrict__ c,
                                              __bf16* __restrict__ hout) {
  __syncthreads();
#pragma unroll
  for (int e = 0; e < 2; ++e) {
    int idx = threadIdx.x + e * 256;       // 512 elements, 256 threads
    int r = idx >> 5, cc = idx & 31;
    int b = m0 + r, n = cg * 32 + cc;
    float gi = gsh[0][r][cc] + bih[n]        + bhh[n];
    float gf = gsh[1][r][cc] + bih[512 + n]  + bhh[512 + n];
    float gg = gsh[2][r][cc] + bih[1024 + n] + bhh[1024 + n];
    float go = gsh[3][r][cc] + bih[1536 + n] + bhh[1536 + n];
    int i = b * H_ + n;
    float cn = sigm(gf) * c[i] + sigm(gi) * tanhf(gg);
    float hn = sigm(go) * tanhf(cn);
    c[i] = cn;
    hout[i] = (__bf16)hn;
  }
}

// ---------------- fused LSTM cell 1: gates GEMM (x=[emb|ctx], h1) + activation --------
// Block = 8 waves: wave w -> gate (w>>1), column half (w&1); all four gates of the
// same 16x32 output slice live in this block, so activation happens in-LDS.
__global__ __launch_bounds__(256) void k_lstm1_fused(int t, const int* __restrict__ y,
                                                     const float* __restrict__ embW,
                                                     const float* __restrict__ ctx,
                                                     const __bf16* __restrict__ h1bf,
                                                     const __bf16* __restrict__ wih,
                                                     const __bf16* __restrict__ whh,
                                                     const float* __restrict__ bih,
                                                     const float* __restrict__ bhh,
                                                     float* __restrict__ c1,
                                                     __bf16* __restrict__ h1out) {
  __shared__ float gsh[4][16][32];
  int lane = threadIdx.x & 31, wave = threadIdx.x >> 5;
  int mtile = blockIdx.x >> 4;           // 0..3  (batch tile)
  int cg    = blockIdx.x & 15;           // 0..15 (32-column group of H)
  int gate  = wave >> 1, sub = wave & 1;
  int m0 = mtile * 16;
  int n0 = gate * 512 + cg * 32 + sub * 16;
  int arow = m0 + (lane & 15);
  int brow = n0 + (lane & 15);
  int tok = (t == 0) ? 0 : y[arow * T_ + t - 1];
  const float*  embrow = embW + (size_t)tok * E_;
  const float*  ctxrow = ctx + arow * P_;
  const __bf16* wb     = wih + (size_t)brow * (E_ + P_);
  const __bf16* hb     = whh + (size_t)brow * H_;
  const __bf16* harow  = h1bf + (size_t)arow * H_;
  v8f acc = {0, 0, 0, 0, 0, 0, 0, 0};
#pragma unroll 1
  for (int k0 = 0; k0 < E_ + P_; k0 += 32) {   // x part (K=768)
    v16bf a = (k0 < E_) ? fragA_f32(embrow, k0, lane)
                        : fragA_f32(ctxrow, k0 - E_, lane);
    acc = WMMA_BF16(a, fragB_bf16(wb, k0, lane), acc);
  }
#pragma unroll 1
  for (int k0 = 0; k0 < H_; k0 += 32) {        // h part (K=512)
    acc = WMMA_BF16(fragA_bf16(harow, k0, lane), fragB_bf16(hb, k0, lane), acc);
  }
  int cloc = (lane & 15) + sub * 16;
  int rb = (lane >> 4) << 3;
#pragma unroll
  for (int r = 0; r < 8; ++r) gsh[gate][rb + r][cloc] = acc[r];
  lstm_epilogue(gsh, m0, cg, bih, bhh, c1, h1out);
}

// ---------------- fused LSTM cell 2 ----------------
__global__ __launch_bounds__(256) void k_lstm2_fused(const __bf16* __restrict__ h1bf,
                                                     const __bf16* __restrict__ h2bf,
                                                     const __bf16* __restrict__ wih,
                                                     const __bf16* __restrict__ whh,
                                                     const float* __restrict__ bih,
                                                     const float* __restrict__ bhh,
                                                     float* __restrict__ c2,
                                                     __bf16* __restrict__ h2out) {
  __shared__ float gsh[4][16][32];
  int lane = threadIdx.x & 31, wave = threadIdx.x >> 5;
  int mtile = blockIdx.x >> 4;
  int cg    = blockIdx.x & 15;
  int gate  = wave >> 1, sub = wave & 1;
  int m0 = mtile * 16;
  int n0 = gate * 512 + cg * 32 + sub * 16;
  int arow = m0 + (lane & 15);
  int brow = n0 + (lane & 15);
  const __bf16* a1 = h1bf + (size_t)arow * H_;
  const __bf16* a2 = h2bf + (size_t)arow * H_;
  const __bf16* b1 = wih + (size_t)brow * H_;
  const __bf16* b2 = whh + (size_t)brow * H_;
  v8f acc = {0, 0, 0, 0, 0, 0, 0, 0};
#pragma unroll 1
  for (int k0 = 0; k0 < H_; k0 += 32)
    acc = WMMA_BF16(fragA_bf16(a1, k0, lane), fragB_bf16(b1, k0, lane), acc);
#pragma unroll 1
  for (int k0 = 0; k0 < H_; k0 += 32)
    acc = WMMA_BF16(fragA_bf16(a2, k0, lane), fragB_bf16(b2, k0, lane), acc);
  int cloc = (lane & 15) + sub * 16;
  int rb = (lane >> 4) << 3;
#pragma unroll
  for (int r = 0; r < 8; ++r) gsh[gate][rb + r][cloc] = acc[r];
  lstm_epilogue(gsh, m0, cg, bih, bhh, c2, h2out);
}

// ---------------- query = h2 @ Wq^T + bq ----------------
__global__ __launch_bounds__(256) void k_query_gemm(const __bf16* __restrict__ h2bf,
                                                    const __bf16* __restrict__ wq,
                                                    const float* __restrict__ bq,
                                                    float* __restrict__ query,
                                                    __bf16* __restrict__ qc) {
  int lane = threadIdx.x & 31, wave = threadIdx.x >> 5;
  int tile = blockIdx.x * 8 + wave;        // 4 x 16 = 64 tiles
  int m0 = (tile >> 4) * 16, n0 = (tile & 15) * 16;
  const __bf16* arow = h2bf + (size_t)(m0 + (lane & 15)) * H_;
  const __bf16* brow = wq + (size_t)(n0 + (lane & 15)) * H_;
  v8f acc = {0, 0, 0, 0, 0, 0, 0, 0};
#pragma unroll 1
  for (int k0 = 0; k0 < H_; k0 += 32)
    acc = WMMA_BF16(fragA_bf16(arow, k0, lane), fragB_bf16(brow, k0, lane), acc);
  int n = n0 + (lane & 15), mb = m0 + ((lane >> 4) << 3);
  float bias = bq[n];
#pragma unroll
  for (int r = 0; r < 8; ++r) {
    float v = acc[r] + bias;
    query[(mb + r) * P_ + n] = v;
    qc[(size_t)(mb + r) * E_ + n] = (__bf16)v;   // first half of [query|ctx]
  }
}

// ---------------- attention: scores, masked softmax, context (bf16 K/V from L2) -------
__global__ __launch_bounds__(256) void k_attn(const __bf16* __restrict__ key,
                                              const __bf16* __restrict__ val,
                                              const float* __restrict__ query,
                                              const int* __restrict__ lens,
                                              float* __restrict__ ctx,
                                              __bf16* __restrict__ qc,
                                              float* __restrict__ attn_out, int t) {
  __shared__ float q_sh[P_];
  __shared__ float sc[S_];
  __shared__ float red[256];
  int b = blockIdx.x, tid = threadIdx.x;
  int len = lens[b];
  q_sh[tid] = query[b * P_ + tid];
  __syncthreads();

  // scores
  float lmax = -__builtin_inff();
  for (int s = tid; s < S_; s += 256) {
    if (s < len) {
      const __bf16* kr = key + ((size_t)b * S_ + s) * P_;
      float dot = 0.f;
#pragma unroll 2
      for (int p0 = 0; p0 < P_; p0 += 8) {
        v8bf kv = *(const v8bf*)(kr + p0);
#pragma unroll
        for (int j = 0; j < 8; ++j) dot += (float)kv[j] * q_sh[p0 + j];
      }
      dot *= 0.0625f;   // 1/sqrt(P)=1/16
      sc[s] = dot;
      lmax = fmaxf(lmax, dot);
    } else {
      sc[s] = -__builtin_inff();
    }
  }
  red[tid] = lmax;
  __syncthreads();
  for (int o = 128; o; o >>= 1) {
    if (tid < o) red[tid] = fmaxf(red[tid], red[tid + o]);
    __syncthreads();
  }
  float m = red[0];
  __syncthreads();

  float lsum = 0.f;
  for (int s = tid; s < S_; s += 256) {
    float e = __expf(sc[s] - m);   // exp(-inf)=0 for masked
    sc[s] = e;
    lsum += e;
  }
  red[tid] = lsum;
  __syncthreads();
  for (int o = 128; o; o >>= 1) {
    if (tid < o) red[tid] += red[tid + o];
    __syncthreads();
  }
  float inv = 1.f / red[0];

  // context: thread <-> channel p; per iteration all 256 lanes read one contiguous
  // bf16 row of V (512B, coalesced, L2-resident)
  int p = tid;
  float acc = 0.f;
#pragma unroll 4
  for (int s = 0; s < len; ++s) {
    const __bf16* vr = val + ((size_t)b * S_ + s) * P_;
    __builtin_prefetch(vr + 8 * P_, 0, 1);
    acc += sc[s] * (float)vr[p];
  }
  acc *= inv;
  ctx[b * P_ + p] = acc;
  qc[(size_t)b * E_ + P_ + p] = (__bf16)acc;   // second half of [query|ctx]

  if (b == 0) {
    for (int s = tid; s < S_; s += 256)
      attn_out[(size_t)t * S_ + s] = sc[s] * inv;
  }
}

// ---------------- logits = [query|ctx] @ emb_W^T + b_out ----------------
__global__ __launch_bounds__(256) void k_logits_gemm(const __bf16* __restrict__ qc,
                                                     const __bf16* __restrict__ embB,
                                                     const float* __restrict__ bout,
                                                     float* __restrict__ pred, int t) {
  int lane = threadIdx.x & 31, wave = threadIdx.x >> 5;
  int tile = blockIdx.x * 8 + wave;            // 4 x 63 = 252 tiles
  const int ntiles = VPAD_ / 16;               // 63
  if (tile >= 4 * ntiles) return;
  int m0 = (tile / ntiles) * 16;
  int n0 = (tile % ntiles) * 16;
  const __bf16* arow = qc + (size_t)(m0 + (lane & 15)) * E_;
  const __bf16* brow = embB + (size_t)(n0 + (lane & 15)) * E_;
  v8f acc = {0, 0, 0, 0, 0, 0, 0, 0};
#pragma unroll 1
  for (int k0 = 0; k0 < E_; k0 += 32)
    acc = WMMA_BF16(fragA_bf16(arow, k0, lane), fragB_bf16(brow, k0, lane), acc);
  int n = n0 + (lane & 15), mb = m0 + ((lane >> 4) << 3);
  if (n < V_) {
    float bias = bout[n];
#pragma unroll
    for (int r = 0; r < 8; ++r)
      pred[(size_t)(mb + r) * T_ * V_ + (size_t)t * V_ + n] = acc[r] + bias;
  }
}

// ---------------- host launch ----------------
extern "C" void kernel_launch(void* const* d_in, const int* in_sizes, int n_in,
                              void* d_out, int out_size, void* d_ws, size_t ws_size,
                              hipStream_t stream) {
  (void)in_sizes; (void)n_in; (void)out_size; (void)ws_size;
  const float* enc  = (const float*)d_in[0];
  const int*   lens = (const int*)d_in[1];
  const int*   y    = (const int*)d_in[2];
  const float* embW = (const float*)d_in[3];
  const float* Wk   = (const float*)d_in[4];
  const float* bk   = (const float*)d_in[5];
  const float* Wv   = (const float*)d_in[6];
  const float* bv   = (const float*)d_in[7];
  const float* Wq   = (const float*)d_in[8];
  const float* bq   = (const float*)d_in[9];
  const float* wih1 = (const float*)d_in[10];
  const float* whh1 = (const float*)d_in[11];
  const float* bih1 = (const float*)d_in[12];
  const float* bhh1 = (const float*)d_in[13];
  const float* wih2 = (const float*)d_in[14];
  const float* whh2 = (const float*)d_in[15];
  const float* bih2 = (const float*)d_in[16];
  const float* bhh2 = (const float*)d_in[17];
  const float* bout = (const float*)d_in[18];

  float* pred = (float*)d_out;
  float* attn = pred + (size_t)B_ * T_ * V_;

  char* ws = (char*)d_ws;
  size_t off = 0;
  auto take = [&](size_t bytes) -> char* {
    char* p = ws + off;
    off += (bytes + 255) & ~(size_t)255;
    return p;
  };
  __bf16* wkB   = (__bf16*)take((size_t)P_ * ENC_ * 2);
  __bf16* wvB   = (__bf16*)take((size_t)P_ * ENC_ * 2);
  __bf16* wqB   = (__bf16*)take((size_t)P_ * H_ * 2);
  __bf16* wih1B = (__bf16*)take((size_t)G_ * (E_ + P_) * 2);
  __bf16* whh1B = (__bf16*)take((size_t)G_ * H_ * 2);
  __bf16* wih2B = (__bf16*)take((size_t)G_ * H_ * 2);
  __bf16* whh2B = (__bf16*)take((size_t)G_ * H_ * 2);
  __bf16* embB  = (__bf16*)take((size_t)VPAD_ * E_ * 2);
  __bf16* keyB  = (__bf16*)take((size_t)B_ * S_ * P_ * 2);
  __bf16* valB  = (__bf16*)take((size_t)B_ * S_ * P_ * 2);
  float*  c1    = (float*)take((size_t)B_ * H_ * 4);
  float*  c2    = (float*)take((size_t)B_ * H_ * 4);
  float*  ctxF  = (float*)take((size_t)B_ * P_ * 4);
  float*  qryF  = (float*)take((size_t)B_ * P_ * 4);
  __bf16* h1B   = (__bf16*)take((size_t)B_ * H_ * 2);
  __bf16* h2B   = (__bf16*)take((size_t)B_ * H_ * 2);
  __bf16* qcB   = (__bf16*)take((size_t)B_ * E_ * 2);

  auto cvt = [&](const float* s, __bf16* d, int n, int ntot) {
    k_f32_to_bf16<<<(ntot + 255) / 256, 256, 0, stream>>>(s, d, n, ntot);
  };
  cvt(Wk, wkB, P_ * ENC_, P_ * ENC_);
  cvt(Wv, wvB, P_ * ENC_, P_ * ENC_);
  cvt(Wq, wqB, P_ * H_, P_ * H_);
  cvt(wih1, wih1B, G_ * (E_ + P_), G_ * (E_ + P_));
  cvt(whh1, whh1B, G_ * H_, G_ * H_);
  cvt(wih2, wih2B, G_ * H_, G_ * H_);
  cvt(whh2, whh2B, G_ * H_, G_ * H_);
  cvt(embW, embB, V_ * E_, VPAD_ * E_);

  k_zero_f32<<<(B_ * H_ + 255) / 256, 256, 0, stream>>>(c1, B_ * H_);
  k_zero_f32<<<(B_ * H_ + 255) / 256, 256, 0, stream>>>(c2, B_ * H_);
  k_zero_f32<<<(B_ * P_ + 255) / 256, 256, 0, stream>>>(ctxF, B_ * P_);
  k_zero_bf16<<<(B_ * H_ + 255) / 256, 256, 0, stream>>>(h1B, B_ * H_);
  k_zero_bf16<<<(B_ * H_ + 255) / 256, 256, 0, stream>>>(h2B, B_ * H_);

  // K/V projection: 6400 m-tiles x 16 n-tiles = 102400 wave-tiles, 8 waves/block
  k_proj<<<12800, 256, 0, stream>>>(enc, wkB, wvB, bk, bv, keyB, valB);

  for (int t = 0; t < T_; ++t) {
    k_lstm1_fused<<<64, 256, 0, stream>>>(t, y, embW, ctxF, h1B, wih1B, whh1B,
                                          bih1, bhh1, c1, h1B);
    k_lstm2_fused<<<64, 256, 0, stream>>>(h1B, h2B, wih2B, whh2B,
                                          bih2, bhh2, c2, h2B);
    k_query_gemm<<<8, 256, 0, stream>>>(h2B, wqB, bq, qryF, qcB);
    k_attn<<<B_, 256, 0, stream>>>(keyB, valB, qryF, lens, ctxF, qcB, attn, t);
    k_logits_gemm<<<32, 256, 0, stream>>>(qcB, embB, bout, pred, t);
  }
}
